// BasicBlock_29523605193207
// MI455X (gfx1250) — 8-bit matrix operand order repaired and checked by reading, compiled, not run on hardware
//
#include <hip/hip_runtime.h>
#include <stdint.h>

typedef __attribute__((ext_vector_type(8))) int v8i;

#define NIMG 32
#define CIN  128
#define HH   56
#define WW   56
#define POUT 128
#define HW   (HH * WW)
#define KSTEPS 18
#define PIX_PER_BLOCK 32
#define TILES_PER_IMG (HW / PIX_PER_BLOCK)
#define NTILES (POUT / 16)

__global__ __launch_bounds__(256) void binact_kernel(const float* __restrict__ x,
                                                     int8_t* __restrict__ act) {
    int pix = blockIdx.x * blockDim.x + threadIdx.x;
    if (pix >= NIMG * HW) return;
    int n  = pix / HW;
    int hw = pix % HW;
    const float* xp = x + (size_t)n * CIN * HW + hw;

    int words[CIN / 4];
#pragma unroll
    for (int i = 0; i < CIN / 4; ++i) {
        int wv = 0;
#pragma unroll
        for (int j = 0; j < 4; ++j) {
            float v = xp[(size_t)(i * 4 + j) * HW];
            int s = (v > 0.f) ? 1 : ((v < 0.f) ? -1 : 0);
            wv |= (s & 0xFF) << (8 * j);
        }
        words[i] = wv;
    }
    int4* dst = (int4*)(act + (size_t)pix * CIN);
    const int* ws = words;
#pragma unroll
    for (int i = 0; i < CIN / 16; ++i)
        dst[i] = make_int4(ws[4 * i], ws[4 * i + 1], ws[4 * i + 2], ws[4 * i + 3]);
}

__global__ __launch_bounds__(256) void binwgt_kernel(const float* __restrict__ w,
                                                     int8_t* __restrict__ bpack) {
    const int TOTAL = NTILES * KSTEPS * 1024;
    int idx = blockIdx.x * blockDim.x + threadIdx.x;
    if (idx >= TOTAL) return;
    int ntile = idx / (KSTEPS * 1024);
    int rem   = idx % (KSTEPS * 1024);
    int ks    = rem / 1024;
    int fb    = rem % 1024;
    int lane  = fb >> 5;
    int vgpr  = (fb >> 2) & 7;
    int byt   = fb & 3;
    int nn    = lane & 15;
    int khalf = lane >> 4;
    int klocal = ((vgpr >> 1) << 4) + (khalf << 3) + ((vgpr & 1) << 2) + byt;
    int tap = ks >> 1, cb = ks & 1;
    int c = cb * 64 + klocal;
    int p = ntile * 16 + nn;
    float v = w[((size_t)p * CIN + c) * 9 + tap];
    bpack[idx] = (int8_t)((v > 0.f) ? 1 : ((v < 0.f) ? -1 : 0));
}

__global__ __launch_bounds__(256) void conv_wmma_kernel(
    const int8_t* __restrict__ act, const int8_t* __restrict__ bpack,
    const float* __restrict__ x, const float* __restrict__ bias,
    const float* __restrict__ gamma, const float* __restrict__ beta,
    const float* __restrict__ rmean, const float* __restrict__ rvar,
    float* __restrict__ out) {
    __shared__ __align__(32) int ldsA[KSTEPS * 2 * 256];

    int blk  = blockIdx.x;
    int n    = blk / TILES_PER_IMG;
    int tile = blk % TILES_PER_IMG;
    int base = tile * PIX_PER_BLOCK;
    int t    = threadIdx.x;

    for (int ci = t; ci < KSTEPS * 2 * 128; ci += 256) {
        int ks = ci >> 8;
        int r  = ci & 255;
        int st = r >> 7;
        int q  = r & 127;
        int lane  = q >> 2;
        int vp    = q & 3;
        int g     = lane >> 4;
        int m     = lane & 15;
        int kbase = vp * 16 + g * 8;
        int tap = ks >> 1, cb = ks & 1;
        int ky = tap / 3, kx = tap % 3;
        int pidx = base + st * 16 + m;
        int oh = pidx / WW, ow = pidx % WW;
        int ih = oh + ky - 1, iw = ow + kx - 1;
        int2 val = make_int2(0, 0);
        if (ih >= 0 && ih < HH && iw >= 0 && iw < WW) {
            val = *(const int2*)(act +
                  ((size_t)n * HW + (size_t)ih * WW + iw) * CIN + cb * 64 + kbase);
        }
        *(int2*)(&ldsA[(ks * 2 + st) * 256 + q * 2]) = val;
    }
    __syncthreads();

    int wave = t >> 5;
    int lane = t & 31;
    const int8_t* bp = bpack + (size_t)wave * KSTEPS * 1024 + (size_t)lane * 32;

    v8i acc0 = {};
    v8i acc1 = {};
    for (int ks = 0; ks < KSTEPS; ++ks) {
        if (ks + 1 < KSTEPS)
            __builtin_prefetch(bp + (ks + 1) * 1024, 0, 1);
        v8i bfrag = *(const v8i*)(bp + ks * 1024);
        v8i a0 = *(const v8i*)(&ldsA[(ks * 2 + 0) * 256 + lane * 8]);
        v8i a1 = *(const v8i*)(&ldsA[(ks * 2 + 1) * 256 + lane * 8]);
        acc0 = __builtin_amdgcn_wmma_i32_16x16x64_iu8(true, a0, true, bfrag, acc0,
                                                      false, false);
        acc1 = __builtin_amdgcn_wmma_i32_16x16x64_iu8(true, a1, true, bfrag, acc1,
                                                      false, false);
    }

    int p  = wave * 16 + (lane & 15);
    int hi = lane >> 4;
    float inv = gamma[p] * rsqrtf(rvar[p] + 1e-5f);
    float sh  = bias[p] * inv + beta[p] - rmean[p] * inv;
    size_t chan = ((size_t)n * POUT + p) * HW;

#pragma unroll
    for (int st = 0; st < 2; ++st) {
        v8i acc = st ? acc1 : acc0;
        int pixBase = base + st * 16 + hi * 8;
        const float4* xr = (const float4*)(x + chan + pixBase);
        float4 xa = xr[0], xb = xr[1];
        float4 oa, ob;
        oa.x = (float)acc[0] * inv + sh + xa.x;
        oa.y = (float)acc[1] * inv + sh + xa.y;
        oa.z = (float)acc[2] * inv + sh + xa.z;
        oa.w = (float)acc[3] * inv + sh + xa.w;
        ob.x = (float)acc[4] * inv + sh + xb.x;
        ob.y = (float)acc[5] * inv + sh + xb.y;
        ob.z = (float)acc[6] * inv + sh + xb.z;
        ob.w = (float)acc[7] * inv + sh + xb.w;
        float4* op = (float4*)(out + chan + pixBase);
        op[0] = oa;
        op[1] = ob;
    }
}

extern "C" void kernel_launch(void* const* d_in, const int* in_sizes, int n_in,
                              void* d_out, int out_size, void* d_ws, size_t ws_size,
                              hipStream_t stream) {
    const float* x      = (const float*)d_in[0];
    const float* weight = (const float*)d_in[1];
    const float* bias   = (const float*)d_in[2];
    const float* gamma  = (const float*)d_in[3];
    const float* beta   = (const float*)d_in[4];
    const float* rmean  = (const float*)d_in[5];
    const float* rvar   = (const float*)d_in[6];
    float* out = (float*)d_out;

    int8_t* act   = (int8_t*)d_ws;
    int8_t* bpack = act + (size_t)NIMG * HW * CIN;

    int actPix = NIMG * HW;
    binact_kernel<<<(actPix + 255) / 256, 256, 0, stream>>>(x, act);

    int wbytes = NTILES * KSTEPS * 1024;
    binwgt_kernel<<<(wbytes + 255) / 256, 256, 0, stream>>>(weight, bpack);

    conv_wmma_kernel<<<NIMG * TILES_PER_IMG, 256, 0, stream>>>(
        act, bpack, x, bias, gamma, beta, rmean, rvar, out);
}
